// Decoder_62191126446120
// MI455X (gfx1250) — matrix operand order repaired and checked by reading, not compiled, not run on hardware
//
#include <hip/hip_runtime.h>
#include <hip/hip_bf16.h>

// ---------------------------------------------------------------------------
// Problem constants (from reference): V=32000, E=512, H=1024, B=64, T=32, S=64
// ---------------------------------------------------------------------------
#define VV     32000
#define EE     512
#define HH     1024
#define BB     64
#define TT     32
#define SS     64
#define KLOG   (EE + HH + 2*HH)   // 3584  logits GEMM K
#define KGI    (EE + 2*HH)        // 2560  gi GEMM K
#define N3H    (3*HH)             // 3072
#define NEG_INF (-1000000000.0f)

typedef __bf16 bf16_t;
typedef __attribute__((ext_vector_type(16))) __bf16 v16bf;
typedef __attribute__((ext_vector_type(8)))  __bf16 v8bf;
typedef __attribute__((ext_vector_type(8)))  float  v8f;

// ---------------------------------------------------------------------------
// WMMA fragment loaders (CDNA5 16x16x32 bf16; contiguous k order, not the ISA table order, same in A and B)
// A (16x32, M x K): lane L holds row M=L%16; elems 0..15 = K[k0+sel*16 ..],
//                   sel = L/16.
// B (32x16, K x N): lane L holds col N=L%16; elems 0..15 = K[k0+sel*16 ..].
//   With BT stored N-major (N x K row-major), each lane load is 32B contiguous.
// ---------------------------------------------------------------------------
__device__ __forceinline__ v16bf load_a_frag(const bf16_t* __restrict__ A,
                                             int K, int m, int k, int sel) {
    return *(const v16bf*)(A + (size_t)m * K + k + sel * 16);
}

__device__ __forceinline__ v16bf load_b_frag(const bf16_t* __restrict__ BT,
                                             int K, int n, int k, int sel) {
    return *(const v16bf*)(BT + (size_t)n * K + k + sel * 16);
}

// ---------------------------------------------------------------------------
// Generic bf16 WMMA GEMM: C[M,N] = A[M,K] * BT[N,K]^T + bias[N]
// grid = (N/64, M/16), block = 128 (4 waves), one 16x16 tile per wave.
// ---------------------------------------------------------------------------
__global__ void gemm_bf16_wmma(const bf16_t* __restrict__ A,
                               const bf16_t* __restrict__ BT,
                               const float*  __restrict__ bias,
                               float* __restrict__ C,
                               int M, int N, int K) {
    const int lane  = threadIdx.x & 31;
    const int wave  = threadIdx.x >> 5;
    const int nt    = blockIdx.x * 4 + wave;
    const int mt    = blockIdx.y;
    if (nt * 16 >= N) return;
    const int laneN = lane & 15;
    const int sel   = lane >> 4;
    const int m     = mt * 16 + laneN;
    const int n     = nt * 16 + laneN;

    v8f acc = {};
    for (int k = 0; k < K; k += 32) {
        v16bf a = load_a_frag(A, K, m, k, sel);
        v16bf b = load_b_frag(BT, K, n, k, sel);
        acc = __builtin_amdgcn_wmma_f32_16x16x32_bf16(
                  false, a, false, b, (short)0, acc, false, false);
    }
    const float bv = bias ? bias[n] : 0.0f;
#pragma unroll
    for (int r = 0; r < 8; ++r) {
        const int row = mt * 16 + r + sel * 8;
        C[(size_t)row * N + n] = acc[r] + bv;
    }
}

// ---------------------------------------------------------------------------
// Logits GEMM: out[b, n] = xo[b, :] (64x3584) . WoT[n, :] + bo[n]
// Wave owns one 16-col tile and all 4 M-tiles (M=64) -> Wo streamed once/step.
// out element written at out[row * rowStride + n] (rowStride = T*V, out
// pre-offset by t*V) to land in the (B,T,V) output.
// grid = (2000/4, 1), block = 128.
// ---------------------------------------------------------------------------
__global__ void logits_gemm_wmma(const bf16_t* __restrict__ A,      // 64 x 3584
                                 const bf16_t* __restrict__ WoT,    // 32000 x 3584
                                 const float*  __restrict__ bo,
                                 float* __restrict__ out,
                                 int rowStride) {
    const int K     = KLOG;
    const int lane  = threadIdx.x & 31;
    const int wave  = threadIdx.x >> 5;
    const int nt    = blockIdx.x * 4 + wave;
    const int laneN = lane & 15;
    const int sel   = lane >> 4;
    const int n     = nt * 16 + laneN;

    v8f acc[4] = {};
    const bf16_t* brow = WoT + (size_t)n * K;
    for (int k = 0; k < K; k += 32) {
        if (k + 512 < K)
            __builtin_prefetch(brow + k + 512, 0, 1);   // global_prefetch_b8
        v16bf b = *(const v16bf*)(brow + k + sel * 16);
#pragma unroll
        for (int mtile = 0; mtile < 4; ++mtile) {
            v16bf a = load_a_frag(A, K, mtile * 16 + laneN, k, sel);
            acc[mtile] = __builtin_amdgcn_wmma_f32_16x16x32_bf16(
                             false, a, false, b, (short)0, acc[mtile], false, false);
        }
    }
    const float bv = bo[n];
#pragma unroll
    for (int mtile = 0; mtile < 4; ++mtile) {
#pragma unroll
        for (int r = 0; r < 8; ++r) {
            const int row = mtile * 16 + r + sel * 8;   // batch index b
            out[(size_t)row * rowStride + n] = acc[mtile][r] + bv;
        }
    }
}

// ---------------------------------------------------------------------------
// f32 -> bf16 conversion (grid-strided)
// ---------------------------------------------------------------------------
__global__ void cvt_bf16_kernel(const float* __restrict__ src,
                                bf16_t* __restrict__ dst, size_t n) {
    size_t i = (size_t)blockIdx.x * blockDim.x + threadIdx.x;
    size_t stride = (size_t)gridDim.x * blockDim.x;
    for (; i < n; i += stride) dst[i] = (bf16_t)src[i];
}

// f32 (K x N) -> bf16 (N x K) transpose-convert (one-time, for Wk)
__global__ void cvt_transpose_kernel(const float* __restrict__ src,
                                     bf16_t* __restrict__ dst,
                                     int K, int N) {
    size_t i = (size_t)blockIdx.x * blockDim.x + threadIdx.x;
    size_t total = (size_t)K * N;
    size_t stride = (size_t)gridDim.x * blockDim.x;
    for (; i < total; i += stride) {
        int k = (int)(i / N);
        int n = (int)(i % N);
        dst[(size_t)n * K + k] = (bf16_t)src[i];
    }
}

// embed gather: emb_bf[b,t,e] = bf16(embed_table[trg[b,t], e])
__global__ void embed_gather_kernel(const int* __restrict__ trg,
                                    const float* __restrict__ table,
                                    bf16_t* __restrict__ emb) {
    size_t i = (size_t)blockIdx.x * blockDim.x + threadIdx.x;
    const size_t total = (size_t)BB * TT * EE;
    if (i >= total) return;
    int e  = (int)(i % EE);
    int bt = (int)(i / EE);                 // b*T + t
    int row = trg[bt];
    emb[i] = (bf16_t)table[(size_t)row * EE + e];
}

// h0 = tanh(encoder_final[-1] @ Wb + bb)   (64 x 1024, K=2048; one-time)
__global__ void h0_kernel(const float* __restrict__ ef,     // B x 2H
                          const float* __restrict__ Wb,     // 2H x H
                          const float* __restrict__ bb,
                          float* __restrict__ h,
                          bf16_t* __restrict__ hbf) {
    int idx = blockIdx.x * blockDim.x + threadIdx.x;
    if (idx >= BB * HH) return;
    int b = idx >> 10, j = idx & (HH - 1);
    float acc = bb[j];
    const float* e = ef + (size_t)b * (2 * HH);
    for (int k = 0; k < 2 * HH; ++k)
        acc = fmaf(e[k], Wb[(size_t)k * HH + j], acc);
    float hv = tanhf(acc);
    h[idx]   = hv;
    hbf[idx] = (bf16_t)hv;
}

// ---------------------------------------------------------------------------
// Per-step attention + context. One block per batch row, 256 threads.
// Fills xg = [e_t | ctx] (bf16, 64x2560) and the e_t / ctx slices of
// xo = [e_t | h_new | ctx] (bf16, 64x3584).
// ---------------------------------------------------------------------------
__global__ void attn_ctx_kernel(const float* __restrict__ h,        // B x H
                                const float* __restrict__ keys,     // B*S x H
                                const float* __restrict__ enc,      // B*S x 2H
                                const unsigned char* __restrict__ mask, // B x S
                                const bf16_t* __restrict__ emb,     // B x T x E
                                int t,
                                bf16_t* __restrict__ xg,
                                bf16_t* __restrict__ xo) {
    __shared__ float s_part[256];
    __shared__ float s_sc[SS];
    __shared__ float s_inv;

    const int b   = blockIdx.x;
    const int tid = threadIdx.x;

    // copy e_t into both concat buffers
    const bf16_t* ep = emb + ((size_t)b * TT + t) * EE;
    for (int i = tid; i < EE; i += 256) {
        bf16_t v = ep[i];
        xg[(size_t)b * KGI  + i] = v;
        xo[(size_t)b * KLOG + i] = v;
    }

    // scores[s] = h[b,:] . keys[b,s,:]   (4 threads per s)
    const int s = tid >> 2;
    const int q = tid & 3;
    const float* hb = h + (size_t)b * HH;
    const float* kp = keys + ((size_t)b * SS + s) * HH;
    float p = 0.0f;
    for (int j = q * 256; j < q * 256 + 256; ++j)
        p = fmaf(hb[j], kp[j], p);
    s_part[tid] = p;
    __syncthreads();
    if (q == 0) {
        float v = s_part[tid] + s_part[tid + 1] + s_part[tid + 2] + s_part[tid + 3];
        s_sc[s] = mask[(size_t)b * SS + s] ? v : NEG_INF;
    }
    __syncthreads();
    if (tid == 0) {                       // softmax over 64 scores (serial, tiny)
        float mx = s_sc[0];
        for (int i = 1; i < SS; ++i) mx = fmaxf(mx, s_sc[i]);
        float sum = 0.0f;
        for (int i = 0; i < SS; ++i) { float e = __expf(s_sc[i] - mx); s_sc[i] = e; sum += e; }
        s_inv = 1.0f / sum;
    }
    __syncthreads();
    const float inv = s_inv;

    // ctx[d] = sum_s attn[s] * enc[b,s,d]
    for (int d = tid; d < 2 * HH; d += 256) {
        const float* row = enc + (size_t)b * SS * (2 * HH) + d;
        float acc = 0.0f;
        for (int ss = 0; ss < SS; ++ss)
            acc = fmaf(s_sc[ss], row[(size_t)ss * (2 * HH)], acc);
        acc *= inv;
        bf16_t v = (bf16_t)acc;
        xg[(size_t)b * KGI  + EE + d]      = v;
        xo[(size_t)b * KLOG + EE + HH + d] = v;
    }
}

// ---------------------------------------------------------------------------
// GRU gate fusion: h_new = (1-z)*n + z*h ; updates h (f32), h_bf16, xo slice
// ---------------------------------------------------------------------------
__global__ void gate_kernel(const float* __restrict__ gi,   // B x 3H
                            const float* __restrict__ gh,   // B x 3H
                            float* __restrict__ h,          // B x H (in/out)
                            bf16_t* __restrict__ hbf,       // B x H
                            bf16_t* __restrict__ xo) {      // B x 3584
    int idx = blockIdx.x * blockDim.x + threadIdx.x;
    if (idx >= BB * HH) return;
    int b = idx >> 10, j = idx & (HH - 1);
    const float* gib = gi + (size_t)b * N3H;
    const float* ghb = gh + (size_t)b * N3H;
    float r  = 1.0f / (1.0f + __expf(-(gib[j]          + ghb[j])));
    float z  = 1.0f / (1.0f + __expf(-(gib[HH + j]     + ghb[HH + j])));
    float nn = tanhf(gib[2 * HH + j] + r * ghb[2 * HH + j]);
    float hv = h[idx];
    float hn = (1.0f - z) * nn + z * hv;
    h[idx]   = hn;
    hbf[idx] = (bf16_t)hn;
    xo[(size_t)b * KLOG + EE + j] = (bf16_t)hn;
}

// ---------------------------------------------------------------------------
// Host-side orchestration
// ---------------------------------------------------------------------------
extern "C" void kernel_launch(void* const* d_in, const int* in_sizes, int n_in,
                              void* d_out, int out_size, void* d_ws, size_t ws_size,
                              hipStream_t stream) {
    const int*           trg   = (const int*)          d_in[0];
    const float*         enc   = (const float*)        d_in[1];   // B,S,2H
    const float*         ef    = (const float*)        d_in[2];   // 1,B,2H
    const unsigned char* mask  = (const unsigned char*)d_in[3];   // B,1,S (bool)
    const float*         table = (const float*)        d_in[4];   // V,E
    const float*         Wk    = (const float*)        d_in[5];   // 2H,H
    const float*         bk    = (const float*)        d_in[6];
    const float*         W_ih  = (const float*)        d_in[7];   // 3H, E+2H
    const float*         W_hh  = (const float*)        d_in[8];   // 3H, H
    const float*         b_ih  = (const float*)        d_in[9];
    const float*         b_hh  = (const float*)        d_in[10];
    const float*         Wb    = (const float*)        d_in[11];  // 2H,H
    const float*         bb    = (const float*)        d_in[12];
    const float*         Wo    = (const float*)        d_in[13];  // V, 3584
    const float*         bo    = (const float*)        d_in[14];
    float*               out   = (float*)d_out;                    // B,T,V

    // -- workspace carve-out (256B aligned) --
    size_t off = 0;
    char* base = (char*)d_ws;
    auto carve = [&](size_t bytes) -> char* {
        char* p = base + off;
        off += (bytes + 255) & ~(size_t)255;
        return p;
    };
    bf16_t* WoB   = (bf16_t*)carve((size_t)VV  * KLOG * 2);   // 229.4 MB
    bf16_t* WihB  = (bf16_t*)carve((size_t)N3H * KGI  * 2);
    bf16_t* WhhB  = (bf16_t*)carve((size_t)N3H * HH   * 2);
    bf16_t* WkT   = (bf16_t*)carve((size_t)HH  * 2*HH * 2);   // H x 2H
    bf16_t* encB  = (bf16_t*)carve((size_t)BB * SS * 2*HH * 2);
    float*  keys  = (float*) carve((size_t)BB * SS * HH * 4);
    bf16_t* embB  = (bf16_t*)carve((size_t)BB * TT * EE * 2);
    float*  hF    = (float*) carve((size_t)BB * HH * 4);
    bf16_t* hB    = (bf16_t*)carve((size_t)BB * HH * 2);
    bf16_t* xg    = (bf16_t*)carve((size_t)BB * KGI  * 2);
    bf16_t* xo    = (bf16_t*)carve((size_t)BB * KLOG * 2);
    float*  gi    = (float*) carve((size_t)BB * N3H * 4);
    float*  gh    = (float*) carve((size_t)BB * N3H * 4);

    // -- one-time precompute --
    cvt_bf16_kernel<<<2048, 256, 0, stream>>>(Wo,   WoB,  (size_t)VV * KLOG);
    cvt_bf16_kernel<<<1024, 256, 0, stream>>>(W_ih, WihB, (size_t)N3H * KGI);
    cvt_bf16_kernel<<<1024, 256, 0, stream>>>(W_hh, WhhB, (size_t)N3H * HH);
    cvt_bf16_kernel<<<1024, 256, 0, stream>>>(enc,  encB, (size_t)BB * SS * 2*HH);
    cvt_transpose_kernel<<<1024, 256, 0, stream>>>(Wk, WkT, 2*HH, HH);
    embed_gather_kernel<<<(BB*TT*EE + 255)/256, 256, 0, stream>>>(trg, table, embB);

    // keys = enc @ Wk + bk : M = B*S = 4096, N = H = 1024, K = 2H = 2048
    gemm_bf16_wmma<<<dim3(HH/64, (BB*SS)/16), 128, 0, stream>>>(
        encB, WkT, bk, keys, BB*SS, HH, 2*HH);

    // h0 = tanh(ef @ Wb + bb)
    h0_kernel<<<(BB*HH + 255)/256, 256, 0, stream>>>(ef, Wb, bb, hF, hB);

    // -- 32 sequential decode steps --
    for (int t = 0; t < TT; ++t) {
        attn_ctx_kernel<<<BB, 256, 0, stream>>>(hF, keys, enc, mask, embB, t, xg, xo);

        // gi = xg @ W_ih^T + b_ih   (64 x 3072, K = 2560)
        gemm_bf16_wmma<<<dim3(N3H/64, BB/16), 128, 0, stream>>>(
            xg, WihB, b_ih, gi, BB, N3H, KGI);
        // gh = h @ W_hh^T + b_hh    (64 x 3072, K = 1024)
        gemm_bf16_wmma<<<dim3(N3H/64, BB/16), 128, 0, stream>>>(
            hB, WhhB, b_hh, gh, BB, N3H, HH);

        gate_kernel<<<(BB*HH + 255)/256, 256, 0, stream>>>(gi, gh, hF, hB, xo);

        // logits[b, t, :] = xo @ Wo^T + bo   (rowStride = T*V)
        logits_gemm_wmma<<<(VV/16)/4, 128, 0, stream>>>(
            xo, WoB, bo, out + (size_t)t * VV, TT * VV);
    }
}
